// AttentionChannel_46385646797621
// MI455X (gfx1250) — compile-verified, matrix operand order checked by reading, not run on hardware
//
#include <hip/hip_runtime.h>
#include <stdint.h>

#define B_      4
#define C_      256
#define HEADS_  8
#define CH_     32
#define HW_     128
#define N_      16384   // 128*128

typedef float  v2f  __attribute__((ext_vector_type(2)));
typedef float  v8f  __attribute__((ext_vector_type(8)));
typedef __bf16 v16bf __attribute__((ext_vector_type(16)));

union ABf { v16bf v; uint4 q[2]; };

__device__ __forceinline__ unsigned short f2bf(float f) {
  unsigned u = __builtin_bit_cast(unsigned, f);
  u += 0x7fffu + ((u >> 16) & 1u);           // round-to-nearest-even
  return (unsigned short)(u >> 16);
}

__device__ __forceinline__ void load_row18(float (&row)[18], const float* xc, int r, int col0) {
  if (r < 0 || r >= HW_) {
    #pragma unroll
    for (int j = 0; j < 18; ++j) row[j] = 0.f;
    return;
  }
  const float* p = xc + r * HW_;
  #pragma unroll
  for (int j = 0; j < 18; ++j) {
    int col = col0 - 1 + j;
    row[j] = (col >= 0 && col < HW_) ? p[col] : 0.f;
  }
}

// -----------------------------------------------------------------------------
// Kernel 1: fused depthwise 3x3 conv (q,k,v) + partial Gram matrix q*k^T.
// grid = B*HEADS*16 row-tiles, block = 256 threads (8 waves).
// thread -> (channel c = tid>>3, col group cg = tid&7, 16 cols each), 8 rows.
// v written to HBM as bf16; q,k staged in LDS and consumed by
// v_wmma_f32_16x16x4_f32 (fp32-exact reduction over pixels).
// -----------------------------------------------------------------------------
__global__ __launch_bounds__(256) void qkv_attn_kernel(
    const float* __restrict__ x,
    const float* __restrict__ qw, const float* __restrict__ qb,
    const float* __restrict__ kw, const float* __restrict__ kb,
    const float* __restrict__ vw, const float* __restrict__ vb,
    unsigned short* __restrict__ vbf,
    float* __restrict__ attn_part)
{
  __shared__ float qs[CH_][132];   // padded: stride 132 words avoids bank conflicts
  __shared__ float ks[CH_][132];
  __shared__ float red[4][256];    // khalf=1 partial accumulators

  const int blk = blockIdx.x;
  const int rt  = blk & 15;
  const int h   = (blk >> 4) & 7;
  const int b   = blk >> 7;
  const int tid = threadIdx.x;
  const int c   = tid >> 3;
  const int cg  = tid & 7;
  const int ch  = h * CH_ + c;
  const int col0 = cg * 16;
  const int r0 = rt * 8;

  float wq[9], wk[9], wv[9];
  #pragma unroll
  for (int i = 0; i < 9; ++i) {
    wq[i] = qw[ch * 9 + i];
    wk[i] = kw[ch * 9 + i];
    wv[i] = vw[ch * 9 + i];
  }
  const float bqv = qb[ch], bkv = kb[ch], bvv = vb[ch];
  const float* xc = x + (size_t)(b * C_ + ch) * N_;

  float rowA[18], rowB[18], rowC[18];
  load_row18(rowA, xc, r0 - 1, col0);
  load_row18(rowB, xc, r0,     col0);

  // WMMA role of this wave: 2x2 grid of 16x16 attn tiles, K split in halves.
  const int wave  = tid >> 5;
  const int lane  = tid & 31;
  const int tile  = wave & 3;
  const int ti    = (tile >> 1) * 16;       // row tile base (c)
  const int tj    = (tile & 1) * 16;        // col tile base (d)
  const int khalf = wave >> 2;
  const int mrow  = lane & 15;
  const int koffA = (lane < 16) ? 0 : 2;    // A 16x4 f32: VGPR0=K0|K2, VGPR1=K1|K3
  v8f acc = {};

  for (int r = 0; r < 8; ++r) {
    load_row18(rowC, xc, r0 + r + 1, col0);

    float qv[16], kv[16], vv[16];
    #pragma unroll
    for (int j = 0; j < 16; ++j) {
      float q = bqv, k = bkv, v = bvv;
      #pragma unroll
      for (int dx = 0; dx < 3; ++dx) {
        float xa = rowA[j + dx], xb = rowB[j + dx], xcv = rowC[j + dx];
        q += xa * wq[dx]     + xb * wq[3 + dx] + xcv * wq[6 + dx];
        k += xa * wk[dx]     + xb * wk[3 + dx] + xcv * wk[6 + dx];
        v += xa * wv[dx]     + xb * wv[3 + dx] + xcv * wv[6 + dx];
      }
      qv[j] = q; kv[j] = k; vv[j] = v;
    }

    // v -> bf16, 32B vector store
    {
      unsigned vr[8];
      #pragma unroll
      for (int j = 0; j < 8; ++j)
        vr[j] = (unsigned)f2bf(vv[2 * j]) | ((unsigned)f2bf(vv[2 * j + 1]) << 16);
      uint4* dst = (uint4*)(vbf + (size_t)(b * C_ + ch) * N_ + (r0 + r) * HW_ + col0);
      dst[0] = make_uint4(vr[0], vr[1], vr[2], vr[3]);
      dst[1] = make_uint4(vr[4], vr[5], vr[6], vr[7]);
    }

    // q,k -> LDS (float4 stores)
    #pragma unroll
    for (int j = 0; j < 16; j += 4) {
      *(float4*)&qs[c][col0 + j] = make_float4(qv[j], qv[j + 1], qv[j + 2], qv[j + 3]);
      *(float4*)&ks[c][col0 + j] = make_float4(kv[j], kv[j + 1], kv[j + 2], kv[j + 3]);
    }
    __syncthreads();

    // Gram accumulation: attn[c,d] += sum_n q[c,n]*k[d,n], K=128 pixels this row
    #pragma unroll
    for (int kk = 0; kk < 64; kk += 4) {
      const int kb2 = khalf * 64 + kk;
      v2f a, bb;
      a.x  = qs[ti + mrow][kb2 + koffA];
      a.y  = qs[ti + mrow][kb2 + koffA + 1];
      bb.x = ks[tj + mrow][kb2 + koffA];
      bb.y = ks[tj + mrow][kb2 + koffA + 1];
      acc = __builtin_amdgcn_wmma_f32_16x16x4_f32(false, a, false, bb,
                                                  (short)0, acc, false, false);
    }
    __syncthreads();

    #pragma unroll
    for (int j = 0; j < 18; ++j) { rowA[j] = rowB[j]; rowB[j] = rowC[j]; }
  }

  // combine K-halves, emit per-tile partial 32x32 Gram
  if (khalf == 1) {
    #pragma unroll
    for (int i = 0; i < 8; ++i) red[wave - 4][lane * 8 + i] = acc[i];
  }
  __syncthreads();
  if (khalf == 0) {
    const int mhalf = (lane & 16) ? 8 : 0;
    const int nloc  = lane & 15;
    float* dst = attn_part + (size_t)(((b * HEADS_ + h) * 16) + rt) * 1024;
    #pragma unroll
    for (int i = 0; i < 8; ++i) {
      float s = acc[i] + red[wave][lane * 8 + i];
      dst[(ti + mhalf + i) * 32 + tj + nloc] = s;
    }
  }
}

// -----------------------------------------------------------------------------
// Kernel 2: reduce 16 partials, apply temperature, row softmax (32-wide).
// -----------------------------------------------------------------------------
__global__ __launch_bounds__(1024) void softmax_kernel(
    const float* __restrict__ attn_part, const float* __restrict__ temp,
    float* __restrict__ attn)
{
  const int bh = blockIdx.x;          // 0..31
  const int h  = bh & 7;
  const int tid = threadIdx.x;
  const int cc = tid >> 5, d = tid & 31;
  const float* src = attn_part + (size_t)bh * 16 * 1024 + cc * 32 + d;
  float v = 0.f;
  #pragma unroll
  for (int t = 0; t < 16; ++t) v += src[t * 1024];
  v *= temp[h];
  float m = v;
  #pragma unroll
  for (int off = 16; off >= 1; off >>= 1) m = fmaxf(m, __shfl_xor(m, off, 32));
  float e = __expf(v - m);
  float s = e;
  #pragma unroll
  for (int off = 16; off >= 1; off >>= 1) s += __shfl_xor(s, off, 32);
  attn[bh * 1024 + cc * 32 + d] = e / s;
}

// -----------------------------------------------------------------------------
// Kernel 3: M = gamma * fusion_w[:, :256] @ out_w ; bias2 = fusion_b + gamma*fwL@out_b
// -----------------------------------------------------------------------------
__global__ __launch_bounds__(256) void prep_m_kernel(
    const float* __restrict__ fw, const float* __restrict__ ow,
    const float* __restrict__ ob, const float* __restrict__ fb,
    const float* __restrict__ gamma,
    float* __restrict__ Mm, float* __restrict__ bias2)
{
  const int o = blockIdx.x, j = threadIdx.x;
  const float g = gamma[0];
  float acc = 0.f;
  for (int i = 0; i < C_; ++i) acc += fw[o * 512 + i] * ow[i * C_ + j];
  Mm[o * C_ + j] = g * acc;
  if (j == 0) {
    float ba = fb[o];
    for (int i = 0; i < C_; ++i) ba += g * fw[o * 512 + i] * ob[i];
    bias2[o] = ba;
  }
}

// -----------------------------------------------------------------------------
// Kernel 4: per-batch combined weight Wcat[b] = [ M*blockdiag(attn_b) | fw_right ]
// converted to bf16. grid = (256 o, 4 b), block = 512 (one thread per k).
// -----------------------------------------------------------------------------
__global__ __launch_bounds__(512) void prep_wcat_kernel(
    const float* __restrict__ Mm, const float* __restrict__ attn,
    const float* __restrict__ fw, unsigned short* __restrict__ Wcat)
{
  const int o = blockIdx.x, b = blockIdx.y, j = threadIdx.x;
  float val;
  if (j < C_) {
    const int h = j >> 5, d = j & 31;
    const float* Mp = Mm + o * C_ + h * CH_;
    const float* ap = attn + (size_t)(b * HEADS_ + h) * 1024 + d;
    float acc = 0.f;
    #pragma unroll
    for (int ci = 0; ci < CH_; ++ci) acc += Mp[ci] * ap[ci * 32];
    val = acc;
  } else {
    val = fw[o * 512 + j];
  }
  Wcat[(size_t)(b * C_ + o) * 512 + j] = f2bf(val);
}

// -----------------------------------------------------------------------------
// Kernel 5: final = Wcat[b] @ [v; x] + bias2, per batch [256x512]x[512x16384].
// bf16 WMMA 16x16x32, f32 accumulate. Block: 64 N-cols x 256 M-rows,
// 8 waves each own 32 M rows -> 2x4 tiles of 16x16.
// -----------------------------------------------------------------------------
__global__ __launch_bounds__(256) void gemm_kernel(
    const unsigned short* __restrict__ Wcat,
    const unsigned short* __restrict__ vbf,
    const float* __restrict__ x,
    const float* __restrict__ bias2,
    float* __restrict__ out)
{
  __shared__ unsigned short Bt[64][40];   // transposed B tile [N][K], padded

  const int b    = blockIdx.y;
  const int n0   = blockIdx.x * 64;
  const int tid  = threadIdx.x;
  const int wave = tid >> 5;
  const int lane = tid & 31;
  const int mrow = lane & 15;
  const int koff = (lane & 16) ? 8 : 0;   // K-half split across lane halves

  const int kk = tid & 31;                // staging: thread -> (k, n-group)
  const int ng = tid >> 5;

  const unsigned short* Ab = Wcat + (size_t)b * C_ * 512;
  v8f acc[2][4] = {};

  for (int kc = 0; kc < 16; ++kc) {
    // stage B chunk: [32 K][64 N] -> Bt[N][K] bf16
    if (kc < 8) {                         // K in [0,256): v (already bf16)
      const unsigned short* src =
          vbf + (size_t)(b * C_ + kc * 32 + kk) * N_ + n0 + ng * 8;
      #pragma unroll
      for (int j = 0; j < 8; ++j) Bt[ng * 8 + j][kk] = src[j];
    } else {                              // K in [256,512): x (fp32 -> bf16)
      const float* src =
          x + (size_t)(b * C_ + (kc - 8) * 32 + kk) * N_ + n0 + ng * 8;
      #pragma unroll
      for (int j = 0; j < 8; ++j) Bt[ng * 8 + j][kk] = f2bf(src[j]);
    }
    __syncthreads();

    ABf afr[2];
    #pragma unroll
    for (int mt = 0; mt < 2; ++mt) {
      const unsigned short* ap =
          Ab + (size_t)(wave * 32 + mt * 16 + mrow) * 512 + kc * 32 + koff;
      afr[mt].q[0] = *(const uint4*)(ap);
      afr[mt].q[1] = *(const uint4*)(ap + 16);
    }
    #pragma unroll
    for (int nt = 0; nt < 4; ++nt) {
      ABf bfr;
      const unsigned short* bp = &Bt[nt * 16 + mrow][koff];
      bfr.q[0] = *(const uint4*)(bp);
      bfr.q[1] = *(const uint4*)(bp + 16);
      #pragma unroll
      for (int mt = 0; mt < 2; ++mt)
        acc[mt][nt] = __builtin_amdgcn_wmma_f32_16x16x32_bf16(
            false, afr[mt].v, false, bfr.v, (short)0, acc[mt][nt], false, false);
    }
    __syncthreads();
  }

  // epilogue: C layout VGPR i -> M = i (lanes 0-15) / i+8 (lanes 16-31), N = lane%16
  const int nlane = lane & 15;
  const int mhalf = (lane & 16) ? 8 : 0;
  #pragma unroll
  for (int mt = 0; mt < 2; ++mt) {
    #pragma unroll
    for (int i = 0; i < 8; ++i) {
      const int m = wave * 32 + mt * 16 + mhalf + i;
      const float bv = bias2[m];
      float* dst = out + (size_t)(b * C_ + m) * N_ + n0;
      #pragma unroll
      for (int nt = 0; nt < 4; ++nt)
        dst[nt * 16 + nlane] = acc[mt][nt][i] + bv;
    }
  }
}

// -----------------------------------------------------------------------------
extern "C" void kernel_launch(void* const* d_in, const int* in_sizes, int n_in,
                              void* d_out, int out_size, void* d_ws, size_t ws_size,
                              hipStream_t stream) {
  (void)in_sizes; (void)n_in; (void)out_size; (void)ws_size;
  const float* x      = (const float*)d_in[0];
  const float* q_w    = (const float*)d_in[1];
  const float* q_b    = (const float*)d_in[2];
  const float* k_w    = (const float*)d_in[3];
  const float* k_b    = (const float*)d_in[4];
  const float* v_w    = (const float*)d_in[5];
  const float* v_b    = (const float*)d_in[6];
  const float* out_w  = (const float*)d_in[7];
  const float* out_b  = (const float*)d_in[8];
  const float* fus_w  = (const float*)d_in[9];
  const float* fus_b  = (const float*)d_in[10];
  const float* temp   = (const float*)d_in[11];
  const float* gamma  = (const float*)d_in[12];

  char* ws = (char*)d_ws;
  unsigned short* vbf   = (unsigned short*)ws;                 // 33,554,432 B
  float* attn_part      = (float*)(ws + 33554432);             //  2,097,152 B
  float* attn           = (float*)(ws + 35651584);             //    131,072 B
  float* Mm             = (float*)(ws + 35782656);             //    262,144 B
  float* bias2          = (float*)(ws + 36044800);             //      1,024 B
  unsigned short* Wcat  = (unsigned short*)(ws + 36045824);    //  1,048,576 B

  qkv_attn_kernel<<<B_ * HEADS_ * 16, 256, 0, stream>>>(
      x, q_w, q_b, k_w, k_b, v_w, v_b, vbf, attn_part);
  softmax_kernel<<<B_ * HEADS_, 1024, 0, stream>>>(attn_part, temp, attn);
  prep_m_kernel<<<C_, 256, 0, stream>>>(fus_w, out_w, out_b, fus_b, gamma, Mm, bias2);
  prep_wcat_kernel<<<dim3(C_, B_), 512, 0, stream>>>(Mm, attn, fus_w, Wcat);
  gemm_kernel<<<dim3(N_ / 64, B_), 256, 0, stream>>>(Wcat, vbf, x, bias2, (float*)d_out);
}
